// TorsionAttention_52999896433139
// MI455X (gfx1250) — matrix operand order repaired and checked by reading, not compiled, not run on hardware
//
#include <hip/hip_runtime.h>
#include <math.h>

typedef __bf16 bf16;
typedef __attribute__((ext_vector_type(16))) __bf16 v16bf;
typedef __attribute__((ext_vector_type(8)))  float  v8f;

#define B_  4
#define S_  2048
#define H_  128
#define NH_ 4
#define HD_ 32

// ---- WMMA fragment helpers (wave32, 16x16x32 bf16) ----

// A-matrix 16x32 (MxK), row-major source with stride ld (bf16 elements).
// Lane layout: lanes0-15 M=lane, elems0-15=K0..15;
//              lanes16-31 M=lane-16, elems0-15=K16..31.
__device__ __forceinline__ v16bf ldA(const bf16* src, int ld) {
  int lane = threadIdx.x & 31;
  int m  = lane & 15;
  int kb = (lane >> 4) << 4;     // 0 or 16
  const bf16* p = src + m * ld + kb;
  v16bf a;
#pragma unroll
  for (int j = 0; j < 8; ++j) a[j]     = p[j];
#pragma unroll
  for (int j = 0; j < 8; ++j) a[8 + j] = p[8 + j];
  return a;
}

// B-matrix 32x16 (KxN) supplied as B^T row-major [N][K] with stride ld.
// Lane layout: lane&15 = N; lanes0-15 K=0..15, lanes16-31 K=16..31.
__device__ __forceinline__ v16bf ldBT(const bf16* srcT, int ld) {
  int lane = threadIdx.x & 31;
  int n  = lane & 15;
  int kb = (lane >> 4) << 4;     // 0 or 16
  const bf16* p = srcT + n * ld + kb;
  v16bf b;
#pragma unroll
  for (int j = 0; j < 16; ++j) b[j] = p[j];
  return b;
}

__device__ __forceinline__ v8f wmma_bf16(v16bf a, v16bf b, v8f c) {
  return __builtin_amdgcn_wmma_f32_16x16x32_bf16(
      /*neg_a=*/false, a, /*neg_b=*/false, b,
      /*c_mod=*/(short)0, c, /*reuse_a=*/false, /*reuse_b=*/false);
}

// C/D 16x16 f32: VGPR j holds row m = j + 8*(lane>>4), col n = lane&15.
__device__ __forceinline__ void stC(float* dst, int ld, v8f c) {
  int lane = threadIdx.x & 31;
  int n  = lane & 15;
  int mh = (lane >> 4) << 3;
#pragma unroll
  for (int j = 0; j < 8; ++j) dst[(mh + j) * ld + n] = c[j];
}

// ======================= Kernel 1: QKV + torsion =======================
// grid 64 (128-row tiles of the 8192 flattened rows), block 256 (8 waves).
__global__ void k_qkv_torsion(const float* __restrict__ x,
                              const float* __restrict__ wq,
                              const float* __restrict__ wk,
                              const float* __restrict__ wv,
                              const float* __restrict__ qta,
                              const float* __restrict__ qtb,
                              const float* __restrict__ kta,
                              const float* __restrict__ ktb,
                              const float* __restrict__ coupling,
                              bf16* __restrict__ qo,
                              bf16* __restrict__ ko,
                              bf16* __restrict__ vo) {
  extern __shared__ char smem[];
  float* acc = (float*)smem;               // [128][132] f32
  float* tas = acc + 128 * 132;            // [2][128][16]
  float* tbs = tas + 2 * 128 * 16;         // [2][16][128]
  float* low = tbs + 2 * 16 * 128;         // [128][2][16]
  bf16*  xs  = (bf16*)(low + 128 * 2 * 16);// [128][136] bf16
  bf16*  wls = xs + 128 * 136;             // [128][136] bf16

  const int t  = threadIdx.x;
  const int w  = t >> 5;
  const int r0 = blockIdx.x * 128;
  const float sig    = 1.0f / (1.0f + __expf(-coupling[0]));
  const float qscale = 0.17677669529663687f;  // HD^-0.5

  for (int i = t; i < 128 * 128; i += 256) {
    int r = i >> 7, c = i & 127;
    xs[r * 136 + c] = (bf16)x[(size_t)(r0 + r) * 128 + c];
  }
  __syncthreads();

  for (int mat = 0; mat < 3; ++mat) {
    const float* wp = (mat == 0) ? wq : ((mat == 1) ? wk : wv);
    for (int i = t; i < 128 * 128; i += 256) {
      int r = i >> 7, c = i & 127;
      wls[r * 136 + c] = (bf16)wp[r * 128 + c];
    }
    if (mat == 0) {
      for (int i = t; i < 4096; i += 256) { tas[i] = qta[i]; tbs[i] = qtb[i]; }
    } else if (mat == 1) {
      for (int i = t; i < 4096; i += 256) { tas[i] = kta[i]; tbs[i] = ktb[i]; }
    }
    __syncthreads();

    // GEMM: wave w produces rows [w*16, w*16+16) x all 128 cols.
    for (int c = 0; c < 8; ++c) {
      v8f cacc = {};
#pragma unroll
      for (int kc = 0; kc < 4; ++kc) {
        v16bf a = ldA(xs + (w * 16) * 136 + kc * 32, 136);
        v16bf b = ldBT(wls + (c * 16) * 136 + kc * 32, 136);
        cacc = wmma_bf16(a, b, cacc);
      }
      stC(acc + (w * 16) * 132 + c * 16, 132, cacc);
    }
    __syncthreads();

    bf16* outp = (mat == 0) ? qo : ((mat == 1) ? ko : vo);
    if (mat < 2) {
      // torsion phase A: low[r][n][0..15] = sum_h acc[r][h] * ta[n][h][j]
      {
        int r = t & 127, n = t >> 7;
        const float* tap = tas + n * 2048;
        float lj[16];
#pragma unroll
        for (int j = 0; j < 16; ++j) lj[j] = 0.f;
        for (int hh = 0; hh < 128; ++hh) {
          float qv = acc[r * 132 + hh];
#pragma unroll
          for (int j = 0; j < 16; ++j) lj[j] += qv * tap[hh * 16 + j];
        }
#pragma unroll
        for (int j = 0; j < 16; ++j) low[r * 32 + n * 16 + j] = lj[j];
      }
      __syncthreads();
      // phase B: lin = low@tb ; corr = sum_n sin(2pi n lin)*lin/n ; store bf16
      {
        int r  = t & 127;
        int h0 = (t >> 7) * 64;
        int rowg = r0 + r;
        int bb = rowg >> 11;
        int ss = rowg & 2047;
        for (int hh = h0; hh < h0 + 64; ++hh) {
          float l1 = 0.f, l2 = 0.f;
#pragma unroll
          for (int j = 0; j < 16; ++j) {
            l1 += low[r * 32 + j]      * tbs[j * 128 + hh];
            l2 += low[r * 32 + 16 + j] * tbs[2048 + j * 128 + hh];
          }
          float corr = __sinf(6.2831853f * l1) * l1 +
                       0.5f * __sinf(12.566371f * l2) * l2;
          float val = acc[r * 132 + hh] + sig * corr;
          if (mat == 0) val *= qscale;
          int hd = hh >> 5, d = hh & 31;
          outp[(((size_t)bb * NH_ + hd) * S_ + ss) * HD_ + d] = (bf16)val;
        }
      }
    } else {
      for (int i = t; i < 128 * 128; i += 256) {
        int r = i >> 7, hh = i & 127;
        int rowg = r0 + r;
        int bb = rowg >> 11, ss = rowg & 2047;
        int hd = hh >> 5, d = hh & 31;
        outp[(((size_t)bb * NH_ + hd) * S_ + ss) * HD_ + d] =
            (bf16)acc[r * 132 + hh];
      }
    }
    __syncthreads();
  }
}

// ======================= Kernel 2: flash attention =======================
// grid 256 = B * NH * (S/128); block 256 (8 waves, 16 q-rows per wave).
__global__ void k_attention(const bf16* __restrict__ qb,
                            const bf16* __restrict__ kb,
                            const bf16* __restrict__ vb,
                            bf16* __restrict__ ao) {
  extern __shared__ char smem[];
  float* sbuf = (float*)smem;              // 8 waves x [16][132] f32
  float* mrow = sbuf + 8 * 16 * 132;       // [128]
  float* lrow = mrow + 128;                // [128]
  float* crow = lrow + 128;                // [128]
  bf16*  qs   = (bf16*)(crow + 128);       // [128][40]
  bf16*  ks   = qs + 128 * 40;             // [128][40]
  bf16*  vts  = ks + 128 * 40;             // [32][136]  (V transposed)
  bf16*  pbuf = vts + 32 * 136;            // 8 waves x [16][136]

  const int t    = threadIdx.x;
  const int w    = t >> 5;
  const int lane = t & 31;
  const int qt = blockIdx.x & 15;
  const int h  = (blockIdx.x >> 4) & 3;
  const int b  = blockIdx.x >> 6;

  const bf16* qhead = qb + (((size_t)b * NH_ + h) * S_) * HD_;
  const bf16* khead = kb + (((size_t)b * NH_ + h) * S_) * HD_;
  const bf16* vhead = vb + (((size_t)b * NH_ + h) * S_) * HD_;

  for (int i = t; i < 128 * 32; i += 256) {
    int r = i >> 5, d = i & 31;
    qs[r * 40 + d] = qhead[(size_t)(qt * 128 + r) * HD_ + d];
  }
  if (lane < 16) { mrow[w * 16 + lane] = -1e30f; lrow[w * 16 + lane] = 0.f; }
  __syncthreads();

  v16bf aq = ldA(qs + (w * 16) * 40, 40);    // HD=32 -> single K-step
  v8f o0 = {}, o1 = {};
  float* sw = sbuf + w * 16 * 132;
  bf16*  pw = pbuf + w * 16 * 136;

  for (int kt = 0; kt < 16; ++kt) {
    __syncthreads();
    for (int i = t; i < 128 * 32; i += 256) {
      int r = i >> 5, d = i & 31;
      ks[r * 40 + d]   = khead[(size_t)(kt * 128 + r) * HD_ + d];
      vts[d * 136 + r] = vhead[(size_t)(kt * 128 + r) * HD_ + d];
    }
    if (kt < 15) __builtin_prefetch(khead + (size_t)(kt + 1) * 128 * HD_, 0, 1);
    __syncthreads();

    // S(16x128) = Q(16x32) * K^T : 8 WMMAs along kv
#pragma unroll
    for (int c = 0; c < 8; ++c) {
      v16bf bk = ldBT(ks + (c * 16) * 40, 40);
      v8f sacc = {};
      sacc = wmma_bf16(aq, bk, sacc);
      stC(sw + c * 16, 132, sacc);
    }
    asm volatile("s_wait_dscnt 0" ::: "memory");

    // online softmax across the 128 columns (2 lanes per row)
    {
      int row  = lane >> 1;
      int half = lane & 1;
      const float* sr = sw + row * 132 + half * 64;
      float mx = -1e30f;
      for (int hh = 0; hh < 64; ++hh) mx = fmaxf(mx, sr[hh]);
      mx = fmaxf(mx, __shfl_xor(mx, 1, 32));
      float oldm = mrow[w * 16 + row];
      float newm = fmaxf(oldm, mx);
      float corr = __expf(oldm - newm);
      float sum = 0.f;
      bf16* pr = pw + row * 136 + half * 64;
      for (int hh = 0; hh < 64; ++hh) {
        float e = __expf(sr[hh] - newm);
        pr[hh] = (bf16)e;
        sum += e;
      }
      sum += __shfl_xor(sum, 1, 32);
      if (half == 0) {
        lrow[w * 16 + row] = lrow[w * 16 + row] * corr + sum;
        mrow[w * 16 + row] = newm;
        crow[w * 16 + row] = corr;
      }
    }
    asm volatile("s_wait_dscnt 0" ::: "memory");

    // rescale O accumulators by per-row correction
    {
      int mh = (lane >> 4) << 3;
#pragma unroll
      for (int j = 0; j < 8; ++j) {
        float cf = crow[w * 16 + mh + j];
        o0[j] *= cf;
        o1[j] *= cf;
      }
    }
    // O(16x32) += P(16x128) * V(128x32) : 4 K-steps x 2 N-subtiles
#pragma unroll
    for (int kc = 0; kc < 4; ++kc) {
      v16bf ap  = ldA(pw + kc * 32, 136);
      v16bf bv0 = ldBT(vts + 0 * 136 + kc * 32, 136);
      v16bf bv1 = ldBT(vts + 16 * 136 + kc * 32, 136);
      o0 = wmma_bf16(ap, bv0, o0);
      o1 = wmma_bf16(ap, bv1, o1);
    }
  }

  // normalize and store merged-head bf16 [B,S,H]
  {
    int n  = lane & 15;
    int mh = (lane >> 4) << 3;
#pragma unroll
    for (int j = 0; j < 8; ++j) {
      int m = mh + j;
      float inv = 1.0f / lrow[w * 16 + m];
      int sp = qt * 128 + w * 16 + m;
      size_t base = ((size_t)b * S_ + sp) * H_ + h * HD_;
      ao[base + n]      = (bf16)(o0[j] * inv);
      ao[base + 16 + n] = (bf16)(o1[j] * inv);
    }
  }
}

// ======================= Kernel 3: output projection =======================
__global__ void k_outproj(const bf16* __restrict__ ain,
                          const float* __restrict__ wo,
                          float* __restrict__ out) {
  extern __shared__ char smem[];
  bf16* as_ = (bf16*)smem;          // [128][136]
  bf16* wls = as_ + 128 * 136;      // [128][136]

  const int t    = threadIdx.x;
  const int w    = t >> 5;
  const int lane = t & 31;
  const int r0   = blockIdx.x * 128;

  for (int i = t; i < 128 * 128; i += 256) {
    int r = i >> 7, c = i & 127;
    as_[r * 136 + c] = ain[(size_t)(r0 + r) * 128 + c];
    wls[r * 136 + c] = (bf16)wo[r * 128 + c];
  }
  __syncthreads();

  int n  = lane & 15;
  int mh = (lane >> 4) << 3;
  for (int c = 0; c < 8; ++c) {
    v8f cacc = {};
#pragma unroll
    for (int kc = 0; kc < 4; ++kc) {
      v16bf a = ldA(as_ + (w * 16) * 136 + kc * 32, 136);
      v16bf b = ldBT(wls + (c * 16) * 136 + kc * 32, 136);
      cacc = wmma_bf16(a, b, cacc);
    }
#pragma unroll
    for (int j = 0; j < 8; ++j)
      out[(size_t)(r0 + w * 16 + mh + j) * 128 + c * 16 + n] = cacc[j];
  }
}

// ======================= launch =======================
extern "C" void kernel_launch(void* const* d_in, const int* in_sizes, int n_in,
                              void* d_out, int out_size, void* d_ws, size_t ws_size,
                              hipStream_t stream) {
  (void)in_sizes; (void)n_in; (void)out_size; (void)ws_size;
  const float* x   = (const float*)d_in[0];
  const float* wq  = (const float*)d_in[1];
  const float* wk  = (const float*)d_in[2];
  const float* wv  = (const float*)d_in[3];
  const float* wo  = (const float*)d_in[4];
  const float* qta = (const float*)d_in[5];
  const float* qtb = (const float*)d_in[6];
  const float* kta = (const float*)d_in[7];
  const float* ktb = (const float*)d_in[8];
  const float* cpl = (const float*)d_in[9];

  char* ws = (char*)d_ws;
  bf16* qbf = (bf16*)(ws);                              // 2 MB  [B,NH,S,HD]
  bf16* kbf = (bf16*)(ws + (size_t)2 * 1024 * 1024);    // 2 MB
  bf16* vbf = (bf16*)(ws + (size_t)4 * 1024 * 1024);    // 2 MB
  bf16* abf = (bf16*)(ws + (size_t)6 * 1024 * 1024);    // 2 MB  [B,S,H]

  size_t sh1 = (size_t)(128 * 132 + 3 * 4096) * 4 + (size_t)(2 * 128 * 136) * 2;
  size_t sh2 = (size_t)(8 * 16 * 132 + 3 * 128) * 4 +
               (size_t)(2 * 128 * 40 + 32 * 136 + 8 * 16 * 136) * 2;
  size_t sh3 = (size_t)(2 * 128 * 136) * 2;

  hipLaunchKernelGGL(k_qkv_torsion, dim3(64), dim3(256), sh1, stream,
                     x, wq, wk, wv, qta, qtb, kta, ktb, cpl, qbf, kbf, vbf);
  hipLaunchKernelGGL(k_attention, dim3(256), dim3(256), sh2, stream,
                     qbf, kbf, vbf, abf);
  hipLaunchKernelGGL(k_outproj, dim3(64), dim3(256), sh3, stream,
                     abf, wo, (float*)d_out);
}
